// GRU_66443144069276
// MI455X (gfx1250) — matrix operand order repaired and checked by reading, not compiled, not run on hardware
//
#include <hip/hip_runtime.h>
#include <hip/hip_bf16.h>

// ---------------------------------------------------------------------------
// GRU (3 layers, B=128, T=512, IN=128, H=1024, PRED=96) on gfx1250.
// bf16 WMMA (v_wmma_f32_16x16x32_bf16) for all matmuls, fp32 accum/state.
// Recurrent w_hh tile is LDS-resident (320KB/WGP) across all 512 timesteps.
// sched_group_barrier forces per-chunk shape: 16 ds_loads, then 8 WMMAs.
// ---------------------------------------------------------------------------

#define Bx    128
#define Tx    512
#define INx   128
#define Hx    1024
#define H3x   3072
#define PREDx 96
#define NBLK_REC 24     // 3072 / 128 persistent blocks for recurrence
#define BPAD  40        // padded LDS row stride in halfs: 80B (16B-aligned, bank step 20)

typedef __attribute__((ext_vector_type(8)))  float   v8f;
typedef __attribute__((ext_vector_type(8)))  __bf16  v8bf;
typedef __attribute__((ext_vector_type(16))) __bf16  v16bf;

__device__ __forceinline__ v16bf cat16(v8bf lo, v8bf hi) {
  return __builtin_shufflevector(lo, hi, 0,1,2,3,4,5,6,7,8,9,10,11,12,13,14,15);
}
__device__ __forceinline__ v8bf lo8(v16bf v) {
  return __builtin_shufflevector(v, v, 0,1,2,3,4,5,6,7);
}
__device__ __forceinline__ v8bf hi8(v16bf v) {
  return __builtin_shufflevector(v, v, 8,9,10,11,12,13,14,15);
}

// ---- WMMA fragment loads (per ISA 7.12.2 16-bit layouts, wave32) ----------
__device__ __forceinline__ v16bf load_a_frag(const __bf16* A, int lda,
                                             int mBase, int k0, int lane) {
  // A-matrix 16x32: lane<16 holds K {k0..k0+7, k0+16..k0+23}; lane>=16 the +8 halves
  int row  = mBase + (lane & 15);
  int koff = ((lane >> 4) & 1) * 8;
  const __bf16* p = A + (size_t)row * lda + k0 + koff;
  v8bf lo = *(const v8bf*)p;
  v8bf hi = *(const v8bf*)(p + 16);
  return cat16(lo, hi);
}

// B fragment from an LDS chunk laid out [row 0..127][BPAD halfs] (k-chunk of 32)
__device__ __forceinline__ v16bf lds_b_frag(const __bf16* buf, int n, int lane) {
  const __bf16* p = buf + (n * 16 + (lane & 15)) * BPAD + ((lane >> 4) & 1) * 8;
  v8bf lo = *(const v8bf*)p;
  v8bf hi = *(const v8bf*)(p + 16);
  return cat16(lo, hi);
}

// 8 WMMAs over one k-chunk: batch all 16 ds_loads, then stream 8 WMMAs.
// sched_group_barrier pins this shape against the register-pressure scheduler.
__device__ __forceinline__ void wmma_chunk(v8f* acc, v16bf a,
                                           const __bf16* buf, int lane) {
  v16bf b[8];
#pragma unroll
  for (int n = 0; n < 8; ++n) b[n] = lds_b_frag(buf, n, lane);
#pragma unroll
  for (int n = 0; n < 8; ++n)
    acc[n] = __builtin_amdgcn_wmma_f32_16x16x32_bf16(
        false, a, false, b[n], (short)0, acc[n], false, false);
#if __has_builtin(__builtin_amdgcn_sched_group_barrier)
  __builtin_amdgcn_sched_group_barrier(0x100, 16, 0);  // 16 DS reads first
  __builtin_amdgcn_sched_group_barrier(0x008, 8, 0);   // then 8 WMMAs back-to-back
#endif
}

// C/D layout: lane L, VGPR r -> (M = mBase + r + 8*(L>=16), N = L%16)
__device__ __forceinline__ void store_tile_bias(const v8f* acc,
                                                const float* __restrict__ bias,
                                                int nBaseGlobal,
                                                float* __restrict__ C, int ldc,
                                                int mBase, int lane) {
  const int colLane = lane & 15;
  const int rowOff  = ((lane >> 4) & 1) * 8;
#pragma unroll
  for (int n = 0; n < 8; ++n) {
    const int col = n * 16 + colLane;
    const float bv = bias ? bias[nBaseGlobal + col] : 0.f;
#pragma unroll
    for (int r = 0; r < 8; ++r) {
      const int row = mBase + rowOff + r;
      C[(size_t)row * ldc + col] = acc[n][r] + bv;
    }
  }
}

// ---- 128x128 GEMM tile with double-buffered LDS staging of B --------------
// C = A(128xK,bf16) * W(128xK,bf16 row-major)^T + bias.  smem: 2*128*BPAD halfs.
__device__ __forceinline__ void gemm_tile_128x128_pipe(
    const __bf16* __restrict__ A, int lda,
    const __bf16* __restrict__ W, int ldb, int K,
    const float* __restrict__ bias, int nBaseGlobal,
    float* __restrict__ C, int ldc, __bf16* smem) {
  const int tid   = threadIdx.x;
  const int lane  = tid & 31;
  const int wave  = tid >> 5;       // 0..7
  const int mBase = wave * 16;
  const int ldRow  = tid >> 1;      // 0..127 (staging row)
  const int ldHalf = (tid & 1) * 16;
  const int KT = K >> 5;

  v8f zero = {0.f,0.f,0.f,0.f,0.f,0.f,0.f,0.f};
  v8f acc[8];
#pragma unroll
  for (int n = 0; n < 8; ++n) acc[n] = zero;

  // stage k-chunk 0 into buffer 0
  v16bf stage = *(const v16bf*)(W + (size_t)ldRow * ldb + ldHalf);
  {
    __bf16* d = smem + ldRow * BPAD + ldHalf;
    *(v8bf*)d       = lo8(stage);
    *(v8bf*)(d + 8) = hi8(stage);
  }
  __syncthreads();

  v16bf a = load_a_frag(A, lda, mBase, 0, lane);
  for (int kt = 0; kt < KT; ++kt) {
    const bool more = (kt + 1) < KT;
    if (more)
      stage = *(const v16bf*)(W + (size_t)ldRow * ldb + ((kt + 1) << 5) + ldHalf);
    v16bf anext = more ? load_a_frag(A, lda, mBase, (kt + 1) << 5, lane) : a;

    const __bf16* buf = smem + (kt & 1) * (128 * BPAD);
    wmma_chunk(acc, a, buf, lane);

    if (more) {
      __bf16* d = smem + ((kt + 1) & 1) * (128 * BPAD) + ldRow * BPAD + ldHalf;
      *(v8bf*)d       = lo8(stage);
      *(v8bf*)(d + 8) = hi8(stage);
    }
    a = anext;
    __syncthreads();
  }
  store_tile_bias(acc, bias, nBaseGlobal, C, ldc, mBase, lane);
}

// ---- device-wide barrier (24 resident blocks, generation counter) ---------
__device__ __forceinline__ void grid_sync(unsigned* sync, unsigned nblk) {
  __syncthreads();
  if (threadIdx.x == 0) {
    __threadfence();
    volatile unsigned* bar = sync;      // generation
    unsigned* cnt = sync + 1;           // arrival count
    unsigned gen = *bar;
    __threadfence();
    if (atomicAdd(cnt, 1u) == nblk - 1u) {
      *cnt = 0u;
      __threadfence();
      atomicAdd((unsigned*)sync, 1u);   // bump generation -> release
    } else {
      while (*bar == gen) { __builtin_amdgcn_s_sleep(1); }
    }
    __threadfence();
  }
  __syncthreads();
}

// ---------------------------------------------------------------------------
// Kernels
// ---------------------------------------------------------------------------

__global__ void f32_to_bf16_kernel(const float* __restrict__ in,
                                   __bf16* __restrict__ out, size_t n) {
  size_t i = blockIdx.x * (size_t)blockDim.x + threadIdx.x;
  size_t stride = gridDim.x * (size_t)blockDim.x;
  for (; i < n; i += stride) out[i] = (__bf16)in[i];
}

// x [B,T,IN] fp32 -> seq [(t*B+b), IN] bf16
__global__ void transpose_x_kernel(const float* __restrict__ x,
                                   __bf16* __restrict__ seq) {
  size_t n = (size_t)Bx * Tx * INx;
  size_t stride = gridDim.x * (size_t)blockDim.x;
  for (size_t i = blockIdx.x * (size_t)blockDim.x + threadIdx.x; i < n; i += stride) {
    size_t ii = i & (INx - 1);
    size_t t  = (i >> 7) & (Tx - 1);
    size_t b  = i >> 16;               // 512*128 = 2^16
    seq[((t * Bx) + b) * INx + ii] = (__bf16)x[i];
  }
}

__global__ void init_state_kernel(float* __restrict__ h,
                                  __bf16* __restrict__ h_bf,
                                  unsigned* __restrict__ sync) {
  int i = blockIdx.x * blockDim.x + threadIdx.x;
  if (i < 2) sync[i] = 0u;
  int stride = gridDim.x * blockDim.x;
  for (int idx = i; idx < Bx * Hx; idx += stride) {
    h[idx] = 0.f;
    h_bf[idx] = (__bf16)0.f;
  }
}

// xg[(t*B+b), 3H] = seq[(t*B+b), K] @ w_ih^T + b_ih   (grid 24 x 512)
__global__ __launch_bounds__(256)
void xg_gemm_kernel(const __bf16* __restrict__ seq, int K,
                    const __bf16* __restrict__ wih,
                    const float* __restrict__ bih,
                    float* __restrict__ xg) {
  __shared__ __bf16 smem[2 * 128 * BPAD];      // 20 KB double buffer
  const int    nBlk = blockIdx.x;              // 0..23
  const size_t mBlk = blockIdx.y;              // 0..511
  const __bf16* A = seq + mBlk * 128 * (size_t)K;
  const __bf16* W = wih + (size_t)(nBlk * 128) * K;
  float* C = xg + mBlk * 128 * (size_t)H3x + nBlk * 128;
  gemm_tile_128x128_pipe(A, K, W, K, K, bih, nBlk * 128, C, H3x, smem);
}

// Persistent recurrence. Each block keeps its 128-row w_hh tile resident in
// LDS (32 k-chunks x 128 rows x BPAD halfs = 320KB) for all T steps.
__global__ __launch_bounds__(256)
void gru_recurrent_kernel(const __bf16* __restrict__ whh,
                          const float* __restrict__ bhh,
                          const float* __restrict__ xg,
                          float* __restrict__ h,
                          __bf16* __restrict__ h_bf,
                          __bf16* __restrict__ yseq,
                          float* __restrict__ hg,
                          unsigned* __restrict__ sync, int T) {
  __shared__ __bf16 wsm[32 * 128 * BPAD];      // 327680 B == 320 KB (full WGP LDS)
  const int nBlk = blockIdx.x;                 // 0..23
  const int tid  = threadIdx.x;
  const int lane = tid & 31;
  const int wave = tid >> 5;
  const int mBase = wave * 16;
  const int gtid = blockIdx.x * blockDim.x + threadIdx.x;
  const int gstride = gridDim.x * blockDim.x;

  // ---- one-time stage: w_hh rows [nBlk*128, +128) x K=1024 into LDS ------
  // 8192 pieces of 16 halfs; fragment-ready layout [kt][row][BPAD]
  for (int idx = tid; idx < 128 * 64; idx += 256) {
    const int row   = idx >> 6;                // 0..127
    const int piece = idx & 63;                // k = piece*16
    const int kt    = piece >> 1;
    const int half  = (piece & 1) * 16;
    v16bf v = *(const v16bf*)(whh + (size_t)(nBlk * 128 + row) * Hx + kt * 32 + half);
    __bf16* d = wsm + ((size_t)kt * 128 + row) * BPAD + half;
    *(v8bf*)d       = lo8(v);
    *(v8bf*)(d + 8) = hi8(v);
  }
  __syncthreads();

  v8f zero = {0.f,0.f,0.f,0.f,0.f,0.f,0.f,0.f};

  for (int t = 0; t < T; ++t) {
    // ---- hg[:, nBlk*128 : +128] = h_bf @ whh^T + bhh (B from LDS) --------
    v8f acc[8];
#pragma unroll
    for (int n = 0; n < 8; ++n) acc[n] = zero;

    v16bf a = load_a_frag(h_bf, Hx, mBase, 0, lane);
    for (int kt = 0; kt < 32; ++kt) {
      v16bf anext = (kt + 1 < 32) ? load_a_frag(h_bf, Hx, mBase, (kt + 1) << 5, lane) : a;
      const __bf16* buf = wsm + (size_t)kt * 128 * BPAD;
      wmma_chunk(acc, a, buf, lane);
      a = anext;
    }
    store_tile_bias(acc, bhh, nBlk * 128, hg + nBlk * 128, H3x, mBase, lane);
    grid_sync(sync, NBLK_REC);

    // ---- gate math over B*H elements, striped across all blocks ----------
    const float* xgt = xg + (size_t)t * Bx * H3x;
    for (int idx = gtid; idx < Bx * Hx; idx += gstride) {
      const int b = idx >> 10;
      const int c = idx & (Hx - 1);
      const size_t rb = (size_t)b * H3x;
      const float xr = xgt[rb + c];
      const float xz = xgt[rb + Hx + c];
      const float xn = xgt[rb + 2 * Hx + c];
      const float hr = hg[rb + c];
      const float hz = hg[rb + Hx + c];
      const float hn = hg[rb + 2 * Hx + c];
      const float r  = 1.f / (1.f + __expf(-(xr + hr)));
      const float z  = 1.f / (1.f + __expf(-(xz + hz)));
      const float nn = tanhf(xn + r * hn);
      const float hnew = (1.f - z) * nn + z * h[idx];
      h[idx]    = hnew;
      const __bf16 hb = (__bf16)hnew;
      h_bf[idx] = hb;
      yseq[((size_t)t * Bx + b) * Hx + c] = hb;
    }
    grid_sync(sync, NBLK_REC);
  }
}

// out[b,p] = h_last[b,:] . fc_w[p,:] + fc_b[p]
__global__ void fc_kernel(const float* __restrict__ h,
                          const float* __restrict__ w,
                          const float* __restrict__ bias,
                          float* __restrict__ out) {
  int i = blockIdx.x * blockDim.x + threadIdx.x;
  if (i >= Bx * PREDx) return;
  const int b = i / PREDx, p = i % PREDx;
  const float* hr = h + (size_t)b * Hx;
  const float* wr = w + (size_t)p * Hx;
  float s = 0.f;
#pragma unroll 4
  for (int k = 0; k < Hx; k += 4)
    s += hr[k] * wr[k] + hr[k+1] * wr[k+1] + hr[k+2] * wr[k+2] + hr[k+3] * wr[k+3];
  out[i] = s + bias[p];
}

// ---------------------------------------------------------------------------
// Host launcher
// ---------------------------------------------------------------------------
extern "C" void kernel_launch(void* const* d_in, const int* in_sizes, int n_in,
                              void* d_out, int out_size, void* d_ws, size_t ws_size,
                              hipStream_t stream) {
  (void)in_sizes; (void)n_in; (void)out_size; (void)ws_size;

  const float* x = (const float*)d_in[0];
  const float* w_ih[3] = {(const float*)d_in[1], (const float*)d_in[5], (const float*)d_in[9]};
  const float* w_hh[3] = {(const float*)d_in[2], (const float*)d_in[6], (const float*)d_in[10]};
  const float* b_ih[3] = {(const float*)d_in[3], (const float*)d_in[7], (const float*)d_in[11]};
  const float* b_hh[3] = {(const float*)d_in[4], (const float*)d_in[8], (const float*)d_in[12]};
  const float* fc_w = (const float*)d_in[13];
  const float* fc_b = (const float*)d_in[14];

  // ---- workspace carve-up (256-byte aligned) ----
  char* ws = (char*)d_ws;
  size_t off = 0;
  auto carve = [&](size_t bytes) { char* p = ws + off; off = (off + bytes + 255) & ~(size_t)255; return p; };

  unsigned* sync = (unsigned*)carve(256);
  __bf16* wih_bf[3]; __bf16* whh_bf[3];
  wih_bf[0] = (__bf16*)carve((size_t)H3x * INx * 2);
  whh_bf[0] = (__bf16*)carve((size_t)H3x * Hx  * 2);
  for (int l = 1; l < 3; ++l) {
    wih_bf[l] = (__bf16*)carve((size_t)H3x * Hx * 2);
    whh_bf[l] = (__bf16*)carve((size_t)H3x * Hx * 2);
  }
  __bf16* seqA = (__bf16*)carve((size_t)Tx * Bx * Hx * 2);
  __bf16* seqB = (__bf16*)carve((size_t)Tx * Bx * Hx * 2);
  float*  xg   = (float*) carve((size_t)Tx * Bx * H3x * 4);
  float*  h    = (float*) carve((size_t)Bx * Hx * 4);
  __bf16* h_bf = (__bf16*)carve((size_t)Bx * Hx * 2);
  float*  hg   = (float*) carve((size_t)Bx * H3x * 4);

  // ---- 1) convert weights to bf16 ----
  f32_to_bf16_kernel<<<512, 256, 0, stream>>>(w_ih[0], wih_bf[0], (size_t)H3x * INx);
  f32_to_bf16_kernel<<<1024, 256, 0, stream>>>(w_hh[0], whh_bf[0], (size_t)H3x * Hx);
  for (int l = 1; l < 3; ++l) {
    f32_to_bf16_kernel<<<1024, 256, 0, stream>>>(w_ih[l], wih_bf[l], (size_t)H3x * Hx);
    f32_to_bf16_kernel<<<1024, 256, 0, stream>>>(w_hh[l], whh_bf[l], (size_t)H3x * Hx);
  }

  // ---- 2) x [B,T,IN] -> seqA [(t,b), IN] bf16 ----
  transpose_x_kernel<<<2048, 256, 0, stream>>>(x, seqA);

  // ---- 3) layers ----
  const __bf16* seqIn = seqA;
  __bf16* seqOut = seqB;
  int K = INx;
  for (int l = 0; l < 3; ++l) {
    xg_gemm_kernel<<<dim3(H3x / 128, (Tx * Bx) / 128), 256, 0, stream>>>(
        seqIn, K, wih_bf[l], b_ih[l], xg);
    init_state_kernel<<<64, 256, 0, stream>>>(h, h_bf, sync);
    gru_recurrent_kernel<<<NBLK_REC, 256, 0, stream>>>(
        whh_bf[l], b_hh[l], xg, h, h_bf, seqOut, hg, sync, Tx);
    // ping-pong sequence buffers
    const __bf16* nextIn = seqOut;
    seqOut = (nextIn == seqA) ? seqB : seqA;
    seqIn  = nextIn;
    K = Hx;
  }

  // ---- 4) final FC from last hidden state ----
  fc_kernel<<<(Bx * PREDx + 255) / 256, 256, 0, stream>>>(h, fc_w, fc_b, (float*)d_out);
}
